// MultiscaleMoE_15607911153878
// MI455X (gfx1250) — matrix operand order repaired and checked by reading, not compiled, not run on hardware
//
#include <hip/hip_runtime.h>
#include <hip/hip_bf16.h>
#include <cstdint>
#include <cstddef>

// ---------------------------------------------------------------------------
// MultiscaleMoE for MI455X (gfx1250)
//   * all heavy GEMMs via v_wmma_f32_16x16x32_bf16 (compute-bound: ~660 GFLOP
//     vs ~0.5 GB of traffic -> bf16 matrix path is the right ceiling)
//   * tile staging via global_load_async_to_lds_b128 + s_wait_asynccnt
//     (ASYNCcnt double buffering, no VGPR round-trip)
//   * weights pre-transposed to [N][K] so A and B staging are identical
//     contiguous copies (no in-loop transpose, no guarded slow path)
//   * last K-round peeled so the steady-state loop is branch-free
// ---------------------------------------------------------------------------

typedef __attribute__((ext_vector_type(16))) __bf16 v16bf;
typedef __attribute__((ext_vector_type(8)))  float  v8f;

#define B_TOK 16384
#define HD    512
#define H2D   1024
#define NEXP  6          // P*S

// GEMM block tile
#define TM 128
#define TN 128
#define TK 32
#define LSTR 40          // LDS row stride in bf16 (80B rows -> 16B aligned frags)
#define BUF_HALFS (2 * TM * LSTR)          // one double-buffer slice (A+B tiles)

__device__ __forceinline__ void async_ld_b128(uint32_t lds_off, const void* g)
{
    // GLOBAL_LOAD_ASYNC_TO_LDS_B128: VDST = LDS byte offset, VADDR = 64b global
    asm volatile("global_load_async_to_lds_b128 %0, %1, off"
                 :: "v"(lds_off), "v"(g) : "memory");
}

#if __has_builtin(__builtin_amdgcn_s_wait_asynccnt)
#define WAIT_ASYNC(n) __builtin_amdgcn_s_wait_asynccnt(n)
#else
#define WAIT_ASYNC(n) asm volatile("s_wait_asynccnt " #n ::: "memory")
#endif

// ---------------------------------------------------------------------------
// C = epilogue(A[MxK] @ BT[NxK]^T + bias)
//   MODE 0: SiLU -> bf16      MODE 1: gate-weighted fp32 accumulate
//   MODE 2: plain fp32 store
// 8 wave32 waves in a 2x4 grid; each wave owns 64x32 = 8 WMMA accumulators.
// ---------------------------------------------------------------------------
template<int MODE>
__global__ __launch_bounds__(256) void moe_gemm_bf16(
    const __bf16* __restrict__ A,  int lda,    // [M][K]
    const __bf16* __restrict__ BT, int ldbt,   // [N][K]  (pre-transposed)
    void* __restrict__ C, int ldc,
    int M, int N, int K,
    const float* __restrict__ bias,
    const float* __restrict__ gate, int gate_idx, int rpb, int beta)
{
    __shared__ __bf16 smem[2 * BUF_HALFS + 8];   // 2 buffers + 16B trash slot

    const int tid  = threadIdx.x;
    const int lane = tid & 31;
    const int wave = tid >> 5;
    const int wm   = wave >> 2;      // 0..1
    const int wn   = wave & 3;       // 0..3
    const int row0 = blockIdx.y * TM;
    const int col0 = blockIdx.x * TN;
    const int r16  = lane & 15;
    const int hi   = lane >> 4;      // half-wave select (ISA 16-bit layouts)

    // ---- async staging coordinates: 4 x 16B chunks per thread per round ----
    const int srow = tid >> 2;            // 0..63
    const int sc   = (tid & 3) * 8;       // half offset within a K-slice row

    const uint32_t lds0  = (uint32_t)(uintptr_t)(void*)smem;  // LDS byte offset
    const uint32_t trash = lds0 + 2 * BUF_HALFS * 2;

    int ar0 = row0 + srow,  ar1 = row0 + srow + 64;
    int br0 = col0 + srow,  br1 = col0 + srow + 64;
    const bool a0ok = ar0 < M, a1ok = ar1 < M;
    const bool b0ok = br0 < N, b1ok = br1 < N;
    if (!a0ok) ar0 = 0;  if (!a1ok) ar1 = 0;   // clamp source (no OOB reads)
    if (!b0ok) br0 = 0;  if (!b1ok) br1 = 0;

    const __bf16* gA0 = A  + (size_t)ar0 * lda  + sc;
    const __bf16* gA1 = A  + (size_t)ar1 * lda  + sc;
    const __bf16* gB0 = BT + (size_t)br0 * ldbt + sc;
    const __bf16* gB1 = BT + (size_t)br1 * ldbt + sc;

    const uint32_t lA0 = lds0 + (srow * LSTR + sc) * 2;
    const uint32_t lA1 = lA0 + 64 * LSTR * 2;
    const uint32_t lB0 = lds0 + (TM * LSTR + srow * LSTR + sc) * 2;
    const uint32_t lB1 = lB0 + 64 * LSTR * 2;
    const uint32_t dA0 = a0ok ? lA0 : trash, dA1 = a1ok ? lA1 : trash;
    const uint32_t dB0 = b0ok ? lB0 : trash, dB1 = b1ok ? lB1 : trash;

    const int R = K / TK;

    auto stage = [&](int r) {
        const int      koff = r * TK;                         // halfs
        const uint32_t bo   = (r & 1) ? (uint32_t)(BUF_HALFS * 2) : 0u;
        async_ld_b128(dA0 + bo, gA0 + koff);
        async_ld_b128(dA1 + bo, gA1 + koff);
        async_ld_b128(dB0 + bo, gB0 + koff);
        async_ld_b128(dB1 + bo, gB1 + koff);
    };

    v8f acc[4][2] = {};

    auto compute_step = [&](int r) {
        const __bf16* Asb = smem + (r & 1) * BUF_HALFS;
        const __bf16* Bsb = Asb + TM * LSTR;
        v16bf afrag[4];
        #pragma unroll
        for (int i = 0; i < 4; ++i) {
            // A 16x32: lanes 0-15 K=0..7 & 16..23, lanes 16-31 K=8..15 & 24..31
            const __bf16* p = &Asb[(wm * 64 + i * 16 + r16) * LSTR + hi * 8];
            union { uint4 u[2]; v16bf v; } f;
            f.u[0] = *(const uint4*)(p);
            f.u[1] = *(const uint4*)(p + 16);
            afrag[i] = f.v;
        }
        v16bf bfrag[2];
        #pragma unroll
        for (int j = 0; j < 2; ++j) {
            // B 32x16: lanes 0-15 K=0..7 & 16..23, lanes 16-31 K=8..15 & 24..31
            const __bf16* p = &Bsb[(wn * 32 + j * 16 + r16) * LSTR + hi * 8];
            union { uint4 u[2]; v16bf v; } f;
            f.u[0] = *(const uint4*)(p);
            f.u[1] = *(const uint4*)(p + 16);
            bfrag[j] = f.v;
        }
        #pragma unroll
        for (int i = 0; i < 4; ++i)
            #pragma unroll
            for (int j = 0; j < 2; ++j)
                acc[i][j] = __builtin_amdgcn_wmma_f32_16x16x32_bf16(
                    false, afrag[i], false, bfrag[j], (short)0, acc[i][j], false, false);
    };

    // ---- software pipeline: branch-free steady state, last round peeled ----
    stage(0);
    for (int r = 0; r < R - 1; ++r) {
        stage(r + 1);
        WAIT_ASYNC(4);          // oldest 4 (round r) complete; r+1 in flight
        __syncthreads();
        int rp = (r + 2 < R) ? (r + 2) : (R - 1);   // clamped GL2 prefetch
        __builtin_prefetch(gA0 + rp * TK, 0, 1);
        __builtin_prefetch(gB0 + rp * TK, 0, 1);
        compute_step(r);
        __syncthreads();        // all reads done before buffer is restaged
    }
    WAIT_ASYNC(0);
    __syncthreads();
    compute_step(R - 1);

    // ---- epilogue (C/D layout: VGPR e -> M = e + 8*hi, N = r16) ----
    #pragma unroll
    for (int i = 0; i < 4; ++i) {
        #pragma unroll
        for (int j = 0; j < 2; ++j) {
            int n = col0 + wn * 32 + j * 16 + r16;
            if (n >= N) continue;
            int   mbase = row0 + wm * 64 + i * 16 + hi * 8;
            float bv    = bias ? bias[n] : 0.f;
            #pragma unroll
            for (int e = 0; e < 8; ++e) {
                int m = mbase + e;
                if (m >= M) continue;
                float  v   = acc[i][j][e] + bv;
                size_t idx = (size_t)m * ldc + n;
                if (MODE == 0) {
                    float s = v * (1.f / (1.f + __expf(-v)));      // SiLU
                    ((__bf16*)C)[idx] = (__bf16)s;
                } else if (MODE == 1) {
                    float wv = gate[(size_t)(m / rpb) * NEXP + gate_idx] * v;
                    float* Cf = (float*)C;
                    if (beta) Cf[idx] += wv; else Cf[idx] = wv;
                } else {
                    ((float*)C)[idx] = v;
                }
            }
        }
    }
}

// ---------------------------------------------------------------------------
// concat [rows,H]+[rows,H] -> bf16 [rows,2H]
// ---------------------------------------------------------------------------
__global__ void moe_concat2_bf16(const float* __restrict__ a, const float* __restrict__ b,
                                 __bf16* __restrict__ out, size_t rows)
{
    size_t idx = (size_t)blockIdx.x * blockDim.x + threadIdx.x;
    size_t total = rows * (size_t)H2D;
    if (idx >= total) return;
    size_t r = idx / H2D;
    int    c = (int)(idx % H2D);
    float  v = (c < HD) ? a[r * HD + c] : b[r * HD + (c - HD)];
    out[idx] = (__bf16)v;
}

__global__ void moe_cvt_bf16(const float* __restrict__ in, __bf16* __restrict__ out, size_t n)
{
    size_t idx = (size_t)blockIdx.x * blockDim.x + threadIdx.x;
    if (idx < n) out[idx] = (__bf16)in[idx];
}

// fp32 [K][N] -> bf16 [N][K] (batched over blockIdx.z slices)
__global__ void moe_cvt_bf16_tr(const float* __restrict__ in, __bf16* __restrict__ out,
                                int K, int N)
{
    size_t slice = (size_t)blockIdx.z * K * N;
    int n = blockIdx.x * 32 + (threadIdx.x & 31);
    int k = blockIdx.y * 8  + (threadIdx.x >> 5);
    if (n < N && k < K)
        out[slice + (size_t)n * K + k] = (__bf16)in[slice + (size_t)k * N + n];
}

// ---------------------------------------------------------------------------
// Tiny gate heads: logits -> softmax -> combined weights w[b, P*S]
// ---------------------------------------------------------------------------
__global__ void moe_gate_finalize(const __bf16* __restrict__ hp,   // [B,128]
                                  const __bf16* __restrict__ hg,   // [B,192] (p-major)
                                  const float* __restrict__ pg_w2, const float* __restrict__ pg_b2,
                                  const float* __restrict__ sg_w2, const float* __restrict__ sg_b2,
                                  float* __restrict__ wout, int Bn)
{
    int b = blockIdx.x * blockDim.x + threadIdx.x;
    if (b >= Bn) return;
    float l0 = pg_b2[0], l1 = pg_b2[1];
    const __bf16* h = hp + (size_t)b * 128;
    for (int j = 0; j < 128; ++j) {
        float x = (float)h[j];
        l0 += x * pg_w2[j * 2 + 0];
        l1 += x * pg_w2[j * 2 + 1];
    }
    float mx = fmaxf(l0, l1);
    float e0 = __expf(l0 - mx), e1 = __expf(l1 - mx);
    float inv = 1.f / (e0 + e1);
    float g[2] = { e0 * inv, e1 * inv };

    #pragma unroll
    for (int p = 0; p < 2; ++p) {
        float gl[3] = { sg_b2[p * 3 + 0], sg_b2[p * 3 + 1], sg_b2[p * 3 + 2] };
        const __bf16* hh = hg + (size_t)b * 192 + p * 96;
        for (int j = 0; j < 96; ++j) {
            float x = (float)hh[j];
            const float* ww = sg_w2 + ((size_t)p * 96 + j) * 3;
            gl[0] += x * ww[0]; gl[1] += x * ww[1]; gl[2] += x * ww[2];
        }
        float m  = fmaxf(gl[0], fmaxf(gl[1], gl[2]));
        float x0 = __expf(gl[0] - m), x1 = __expf(gl[1] - m), x2 = __expf(gl[2] - m);
        float iv = 1.f / (x0 + x1 + x2);
        wout[(size_t)b * NEXP + p * 3 + 0] = g[p] * x0 * iv;
        wout[(size_t)b * NEXP + p * 3 + 1] = g[p] * x1 * iv;
        wout[(size_t)b * NEXP + p * 3 + 2] = g[p] * x2 * iv;
    }
}

// ---------------------------------------------------------------------------
// KL load-balance loss (single block reduction)
// ---------------------------------------------------------------------------
__global__ void moe_loss_kernel(const float* __restrict__ w, int Bn, float* __restrict__ out)
{
    __shared__ float sh[NEXP][256];
    float s[NEXP] = {0, 0, 0, 0, 0, 0};
    for (int b = threadIdx.x; b < Bn; b += 256)
        #pragma unroll
        for (int e = 0; e < NEXP; ++e) s[e] += w[(size_t)b * NEXP + e];
    #pragma unroll
    for (int e = 0; e < NEXP; ++e) sh[e][threadIdx.x] = s[e];
    __syncthreads();
    for (int stride = 128; stride > 0; stride >>= 1) {
        if ((int)threadIdx.x < stride)
            #pragma unroll
            for (int e = 0; e < NEXP; ++e) sh[e][threadIdx.x] += sh[e][threadIdx.x + stride];
        __syncthreads();
    }
    if (threadIdx.x == 0) {
        const float ideal = 1.f / (float)NEXP;
        float loss = 0.f;
        #pragma unroll
        for (int e = 0; e < NEXP; ++e) {
            float usage = sh[e][0] / (float)Bn;
            loss += ideal * (logf(ideal) - logf(usage + 1e-10f));
        }
        out[0] = loss / (float)NEXP * 0.05f;
    }
}

// ---------------------------------------------------------------------------
extern "C" void kernel_launch(void* const* d_in, const int* in_sizes, int n_in,
                              void* d_out, int out_size, void* d_ws, size_t ws_size,
                              hipStream_t stream)
{
    const float* scalar_short = (const float*)d_in[0];
    const float* scalar_long  = (const float*)d_in[1];
    const float* vector_short = (const float*)d_in[2];
    const float* vector_long  = (const float*)d_in[3];
    const float* pg_w1 = (const float*)d_in[4];
    const float* pg_b1 = (const float*)d_in[5];
    const float* pg_w2 = (const float*)d_in[6];
    const float* pg_b2 = (const float*)d_in[7];
    const float* sg_w1 = (const float*)d_in[8];
    const float* sg_b1 = (const float*)d_in[9];
    const float* sg_w2 = (const float*)d_in[10];
    const float* sg_b2 = (const float*)d_in[11];
    const float* se_w1 = (const float*)d_in[12];
    const float* se_b1 = (const float*)d_in[13];
    const float* se_w2 = (const float*)d_in[14];
    const float* se_b2 = (const float*)d_in[15];
    const float* ve_w1 = (const float*)d_in[16];
    const float* ve_w2 = (const float*)d_in[17];
    const float* int_s_w = (const float*)d_in[18];
    const float* int_s_b = (const float*)d_in[19];
    const float* int_v_w = (const float*)d_in[20];

    float* out_s    = (float*)d_out;                       // [B, H]
    float* out_v    = out_s + (size_t)B_TOK * HD;          // [B, 3, H]
    float* out_loss = out_v + (size_t)B_TOK * 3 * HD;      // [1]

    // ---- workspace carve-out ----
    char*  ws  = (char*)d_ws;
    size_t off = 0;
    auto alloc = [&](size_t bytes) -> void* {
        void* p = ws + off;
        off = (off + bytes + 255) & ~(size_t)255;
        return p;
    };
    const size_t M3 = (size_t)B_TOK * 3;
    __bf16* sc_bf   = (__bf16*)alloc((size_t)B_TOK * H2D * 2);
    __bf16* vc_bf   = (__bf16*)alloc(M3 * H2D * 2);
    __bf16* w_pg1   = (__bf16*)alloc((size_t)H2D * 128 * 2);        // [128][1024]
    __bf16* w_sg1   = (__bf16*)alloc((size_t)2 * H2D * 96 * 2);     // [p][96][1024]
    __bf16* w_se1   = (__bf16*)alloc((size_t)NEXP * H2D * HD * 2);  // [e][512][1024]
    __bf16* w_se2   = (__bf16*)alloc((size_t)NEXP * HD * HD * 2);   // [e][512][512]
    __bf16* w_ve1   = (__bf16*)alloc((size_t)NEXP * H2D * HD * 2);
    __bf16* w_ve2   = (__bf16*)alloc((size_t)NEXP * HD * HD * 2);
    __bf16* w_ints  = (__bf16*)alloc((size_t)HD * HD * 2);
    __bf16* w_intv  = (__bf16*)alloc((size_t)HD * HD * 2);
    __bf16* hp      = (__bf16*)alloc((size_t)B_TOK * 128 * 2);
    __bf16* hg      = (__bf16*)alloc((size_t)B_TOK * 192 * 2);
    float*  gatew   = (float*) alloc((size_t)B_TOK * NEXP * 4);
    __bf16* eh      = (__bf16*)alloc((size_t)B_TOK * HD * 2);
    __bf16* vh      = (__bf16*)alloc(M3 * HD * 2);
    float*  sacc    = (float*) alloc((size_t)B_TOK * HD * 4);
    float*  vacc    = (float*) alloc(M3 * HD * 4);
    __bf16* sacc_bf = (__bf16*)alloc((size_t)B_TOK * HD * 2);
    __bf16* vacc_bf = (__bf16*)alloc(M3 * HD * 2);

    auto cvt = [&](const float* in, __bf16* outp, size_t n) {
        unsigned blocks = (unsigned)((n + 255) / 256);
        moe_cvt_bf16<<<blocks, 256, 0, stream>>>(in, outp, n);
    };
    auto cvt_tr = [&](const float* in, __bf16* outp, int K, int N, int z) {
        dim3 grid((N + 31) / 32, (K + 7) / 8, z);
        moe_cvt_bf16_tr<<<grid, 256, 0, stream>>>(in, outp, K, N);
    };
    auto gemm = [&](int mode, const __bf16* A, int lda, const __bf16* BTp, int ldbt,
                    void* C, int ldc, int M, int N, int K,
                    const float* bias, const float* gate, int gidx, int rpb, int beta) {
        dim3 grid((N + TN - 1) / TN, (M + TM - 1) / TM);
        if (mode == 0)
            moe_gemm_bf16<0><<<grid, 256, 0, stream>>>(A, lda, BTp, ldbt, C, ldc,
                                                       M, N, K, bias, gate, gidx, rpb, beta);
        else if (mode == 1)
            moe_gemm_bf16<1><<<grid, 256, 0, stream>>>(A, lda, BTp, ldbt, C, ldc,
                                                       M, N, K, bias, gate, gidx, rpb, beta);
        else
            moe_gemm_bf16<2><<<grid, 256, 0, stream>>>(A, lda, BTp, ldbt, C, ldc,
                                                       M, N, K, bias, gate, gidx, rpb, beta);
    };

    // ---- 1. bf16 conversion: activations concatenated, weights transposed ----
    {
        size_t t1 = (size_t)B_TOK * H2D;
        moe_concat2_bf16<<<(unsigned)((t1 + 255) / 256), 256, 0, stream>>>(
            scalar_short, scalar_long, sc_bf, (size_t)B_TOK);
        size_t t2 = M3 * H2D;
        moe_concat2_bf16<<<(unsigned)((t2 + 255) / 256), 256, 0, stream>>>(
            vector_short, vector_long, vc_bf, M3);
    }
    cvt_tr(pg_w1,   w_pg1,  H2D, 128, 1);
    cvt_tr(sg_w1,   w_sg1,  H2D, 96,  2);
    cvt_tr(se_w1,   w_se1,  H2D, HD,  NEXP);
    cvt_tr(se_w2,   w_se2,  HD,  HD,  NEXP);
    cvt_tr(ve_w1,   w_ve1,  H2D, HD,  NEXP);
    cvt_tr(ve_w2,   w_ve2,  HD,  HD,  NEXP);
    cvt_tr(int_s_w, w_ints, HD,  HD,  1);
    cvt_tr(int_v_w, w_intv, HD,  HD,  1);

    // ---- 2. gate hidden layers (WMMA) + tiny heads ----
    gemm(0, sc_bf, H2D, w_pg1, H2D, hp, 128, B_TOK, 128, H2D,
         pg_b1, nullptr, 0, 1, 0);
    for (int p = 0; p < 2; ++p)
        gemm(0, sc_bf, H2D, w_sg1 + (size_t)p * H2D * 96, H2D,
             hg + p * 96, 192, B_TOK, 96, H2D,
             sg_b1 + p * 96, nullptr, 0, 1, 0);
    moe_gate_finalize<<<(B_TOK + 255) / 256, 256, 0, stream>>>(
        hp, hg, pg_w2, pg_b2, sg_w2, sg_b2, gatew, B_TOK);

    // ---- 3. experts (dominant cost, all WMMA) ----
    for (int e = 0; e < NEXP; ++e) {
        int beta = (e > 0);
        // scalar expert
        gemm(0, sc_bf, H2D, w_se1 + (size_t)e * H2D * HD, H2D, eh, HD,
             B_TOK, HD, H2D, se_b1 + (size_t)e * HD, nullptr, 0, 1, 0);
        gemm(1, eh, HD, w_se2 + (size_t)e * HD * HD, HD, sacc, HD,
             B_TOK, HD, HD, se_b2 + (size_t)e * HD, gatew, e, 1, beta);
        // vector expert (rows = B*3, gate indexed by row/3; no biases)
        gemm(0, vc_bf, H2D, w_ve1 + (size_t)e * H2D * HD, H2D, vh, HD,
             (int)M3, HD, H2D, nullptr, nullptr, 0, 1, 0);
        gemm(1, vh, HD, w_ve2 + (size_t)e * HD * HD, HD, vacc, HD,
             (int)M3, HD, HD, nullptr, gatew, e, 3, beta);
    }

    // ---- 4. integration layers ----
    cvt(sacc, sacc_bf, (size_t)B_TOK * HD);
    cvt(vacc, vacc_bf, M3 * HD);
    gemm(2, sacc_bf, HD, w_ints, HD, out_s, HD, B_TOK, HD, HD,
         int_s_b, nullptr, 0, 1, 0);
    gemm(2, vacc_bf, HD, w_intv, HD, out_v, HD, (int)M3, HD, HD,
         nullptr, nullptr, 0, 1, 0);

    // ---- 5. load-balance loss ----
    moe_loss_kernel<<<1, 256, 0, stream>>>(gatew, B_TOK, out_loss);
}
